// MultiHeadAttention_17386027614536
// MI455X (gfx1250) — matrix operand order repaired and checked by reading, not compiled, not run on hardware
//
#include <hip/hip_runtime.h>
#include <hip/hip_bf16.h>

// ---------------------------------------------------------------------------
// Causal MHA forward for MI455X (gfx1250): bf16 WMMA + TDM-staged GEMMs.
//   x:(2,2048,1024) wq/wk/wv/wo:(1024,1024)  ->  out:(2,2048,1024) fp32
// ---------------------------------------------------------------------------

typedef __bf16 bf16;
typedef __attribute__((ext_vector_type(16))) __bf16     v16bf;
typedef __attribute__((ext_vector_type(8)))  float      v8f;
typedef __attribute__((ext_vector_type(4)))  unsigned   v4u;
typedef __attribute__((ext_vector_type(8)))  unsigned   v8u;

union FragBF {          // one WMMA 16x32 bf16 operand: 16 bf16 per lane = 8 VGPRs
    v16bf v;
    v4u   u[2];
};

static __device__ __forceinline__ v8f wmma_bf16(const FragBF& a, const FragBF& b, v8f c) {
    return __builtin_amdgcn_wmma_f32_16x16x32_bf16(false, a.v, false, b.v,
                                                   (short)0, c, false, false);
}

// ---------------------------------------------------------------------------
// Tensor Data Mover: DMA a 2D tile (tile_d1 rows x tile_d0 elems, bf16) from
// global memory into LDS.  Builds the D# (group0 + group1) per CDNA5 ISA
// CDNA5 ISA and issues TENSOR_LOAD_TO_LDS via inline asm (portable
// across ROCm7.2 / clang-23 builtin arity differences).  Tracked by TENSORcnt.
// ---------------------------------------------------------------------------
static __device__ __forceinline__ void tdm_load_2d_bf16(
        unsigned lds_off, const bf16* gptr,
        unsigned tensor_d0, unsigned tensor_d1,
        unsigned tile_d0,   unsigned tile_d1,
        unsigned row_stride /* elements */)
{
    unsigned long long ga = (unsigned long long)(uintptr_t)gptr;
    v4u g0;
    g0[0] = 1u;                                            // count=1, user D#
    g0[1] = lds_off;                                       // lds_addr
    g0[2] = (unsigned)(ga & 0xffffffffu);                  // global_addr[31:0]
    g0[3] = (unsigned)((ga >> 32) & 0x01ffffffu)
          | (2u << 30);                                    // type=2 ("image")
    v8u g1;
    g1[0] = (1u << 16);                                    // data_size=2 bytes
    g1[1] = (tensor_d0 & 0xffffu) << 16;                   // tensor_dim0[15:0]
    g1[2] = (tensor_d0 >> 16) | ((tensor_d1 & 0xffffu) << 16);
    g1[3] = (tensor_d1 >> 16) | (tile_d0 << 16);           // tile_dim0
    g1[4] = tile_d1;                                       // tile_dim1 (dim2=0)
    g1[5] = row_stride;                                    // dim0_stride[31:0]
    g1[6] = 0u;
    g1[7] = 0u;
    asm volatile("tensor_load_to_lds %0, %1" :: "s"(g0), "s"(g1) : "memory");
}

// ---------------------------------------------------------------------------
// fp32 -> bf16 conversion (vector-4)
// ---------------------------------------------------------------------------
__global__ void mha_cvt_bf16(const float* __restrict__ src, bf16* __restrict__ dst, int n4) {
    int i = blockIdx.x * blockDim.x + threadIdx.x;
    if (i >= n4) return;
    float4 f = reinterpret_cast<const float4*>(src)[i];
    union { bf16 e[4]; unsigned long long q; } p;
    p.e[0] = (bf16)f.x; p.e[1] = (bf16)f.y; p.e[2] = (bf16)f.z; p.e[3] = (bf16)f.w;
    reinterpret_cast<unsigned long long*>(dst)[i] = p.q;
}

// ---------------------------------------------------------------------------
// Tiled bf16 GEMM:  C[m][n] = sum_k A[m][k] * B[n][k]      (i.e. A @ B^T)
//   M x 1024, K = 1024.  Block tile 128x128, 8 waves (2x4), wave tile 64x32.
//   K-tiles staged by the Tensor Data Mover, double-buffered in LDS.
// mode 0: fp32 store to Cout[m*1024+n]                      (out-projection)
// mode 1: RoPE + 1/sqrt(dk) scale, bf16 store [b,h,n,64]    (Q)
// mode 2: RoPE,                 bf16 store [b,h,n,64]       (K)
// mode 3:                       bf16 store [b,h,64,n] (V^T) (V)
// ---------------------------------------------------------------------------
__global__ void __launch_bounds__(256)
mha_gemm_bf16(const bf16* __restrict__ A, const bf16* __restrict__ B,
              void* __restrict__ Cout, int K, int mode)
{
    __shared__ __align__(16) bf16 As[2][128][32];
    __shared__ __align__(16) bf16 Bs[2][128][32];

    const int tid  = threadIdx.x;
    const int lane = tid & 31;
    const int wave = tid >> 5;
    const int half = lane >> 4;     // which 16-lane half
    const int l16  = lane & 15;
    const int wm   = (wave >> 2) * 64;   // wave M offset in block tile
    const int wn   = (wave & 3)  * 32;   // wave N offset in block tile
    const int m0   = blockIdx.y * 128;
    const int n0   = blockIdx.x * 128;

    const unsigned ldsA0 = (unsigned)(uintptr_t)&As[0][0][0];
    const unsigned ldsA1 = (unsigned)(uintptr_t)&As[1][0][0];
    const unsigned ldsB0 = (unsigned)(uintptr_t)&Bs[0][0][0];
    const unsigned ldsB1 = (unsigned)(uintptr_t)&Bs[1][0][0];

    v8f acc[4][2];
    #pragma unroll
    for (int s = 0; s < 4; ++s)
        #pragma unroll
        for (int t = 0; t < 2; ++t)
            #pragma unroll
            for (int i = 0; i < 8; ++i) acc[s][t][i] = 0.0f;

    // ---- prologue: DMA first K-tile into buffer 0 (wave 0 only) ----
    if (wave == 0) {
        tdm_load_2d_bf16(ldsA0, A + (size_t)m0 * K, K, 128, 32, 128, K);
        tdm_load_2d_bf16(ldsB0, B + (size_t)n0 * K, K, 128, 32, 128, K);
    }

    int buf = 0;
    for (int k0 = 0; k0 < K; k0 += 32) {
        // ---- issue next tile's DMA, then wait for the current one ----
        if (wave == 0) {
            if (k0 + 32 < K) {
                const unsigned la = buf ? ldsA0 : ldsA1;
                const unsigned lb = buf ? ldsB0 : ldsB1;
                tdm_load_2d_bf16(la, A + (size_t)m0 * K + (k0 + 32), K, 128, 32, 128, K);
                tdm_load_2d_bf16(lb, B + (size_t)n0 * K + (k0 + 32), K, 128, 32, 128, K);
                asm volatile("s_wait_tensorcnt 0x2" ::: "memory");
            } else {
                asm volatile("s_wait_tensorcnt 0x0" ::: "memory");
            }
        }
        __syncthreads();

        // ---- build fragments per ISA VGPR layouts ----
        FragBF af[4], bf[2];
        #pragma unroll
        for (int s = 0; s < 4; ++s) {   // A: lane<16 holds K {0-7,16-23}
            const char* base = (const char*)&As[buf][wm + s * 16 + l16][0];
            af[s].u[0] = *(const v4u*)(base + half * 16);
            af[s].u[1] = *(const v4u*)(base + half * 16 + 32);
        }
        #pragma unroll
        for (int t = 0; t < 2; ++t) {   // B: lane<16 holds K {0-7,16-23}
            const char* base = (const char*)&Bs[buf][wn + t * 16 + l16][0] + half * 16;
            bf[t].u[0] = *(const v4u*)(base);
            bf[t].u[1] = *(const v4u*)(base + 32);
        }
        #pragma unroll
        for (int s = 0; s < 4; ++s)
            #pragma unroll
            for (int t = 0; t < 2; ++t)
                acc[s][t] = wmma_bf16(af[s], bf[t], acc[s][t]);
        __syncthreads();                // all waves done reading before overwrite
        buf ^= 1;
    }

    // ---- epilogue ----
    #pragma unroll
    for (int s = 0; s < 4; ++s) {
        #pragma unroll
        for (int t = 0; t < 2; ++t) {
            #pragma unroll
            for (int i = 0; i < 8; ++i) {
                const int gm = m0 + wm + s * 16 + i + 8 * half;   // token row
                const int gn = n0 + wn + t * 16 + l16;            // feature col
                float val = acc[s][t][i];
                if (mode == 0) {
                    ((float*)Cout)[(size_t)gm * 1024 + gn] = val;
                } else {
                    const int b = gm >> 11, npos = gm & 2047;
                    const int h = gn >> 6,  d    = gn & 63;
                    if (mode == 3) {
                        // V stored transposed per head: [b,h,d,npos]
                        ((bf16*)Cout)[((size_t)((b * 16 + h) * 64 + d)) * 2048 + npos] = (bf16)val;
                    } else {
                        // RoPE: pair partner sits in adjacent lane (col = lane&15)
                        float partner = __shfl_xor(val, 1, 32);
                        // freq = 10000^(-(d&~1)/64) ; ln(10000)/64 = 0.14391156...
                        float freq = __expf(-0.14391156f * (float)(d & 62));
                        float sn, cs;
                        __sincosf((float)npos * freq, &sn, &cs);
                        float r = (d & 1) ? (partner * sn + val * cs)
                                          : (val * cs - partner * sn);
                        if (mode == 1) r *= 0.125f;   // 1/sqrt(64) folded into Q
                        ((bf16*)Cout)[((size_t)((b * 16 + h) * 2048 + npos)) * 64 + d] = (bf16)r;
                    }
                }
            }
        }
    }
}

// ---------------------------------------------------------------------------
// Flash attention (causal), per (b,h): Q,K:[2048,64] bf16 (Q pre-scaled),
// V^T:[64,2048] bf16.  8 waves/block, each wave owns 16 query rows.
// Output O: bf16 [token=4096][1024] feeding the final projection.
// ---------------------------------------------------------------------------
__global__ void __launch_bounds__(256)
mha_flash_attn(const bf16* __restrict__ Qp, const bf16* __restrict__ Kp,
               const bf16* __restrict__ Vtp, bf16* __restrict__ Op)
{
    const int Nn   = 2048;
    const int bh   = blockIdx.y;               // 0..31  (b*16 + h)
    const int lane = threadIdx.x & 31;
    const int wave = threadIdx.x >> 5;
    const int half = lane >> 4;
    const int l16  = lane & 15;
    const int qm   = blockIdx.x * 128 + wave * 16;   // this wave's query base

    __shared__ __align__(16) bf16 Ps[8][16][32];     // per-wave P transpose slab

    const bf16* Qh = Qp  + (size_t)bh * Nn * 64;
    const bf16* Kh = Kp  + (size_t)bh * Nn * 64;
    const bf16* Vh = Vtp + (size_t)bh * 64 * Nn;

    // Q fragments (A operand, two K=32 chunks over d_k=64), loaded once
    FragBF qa[2];
    #pragma unroll
    for (int c = 0; c < 2; ++c) {
        const char* base = (const char*)(Qh + (size_t)(qm + l16) * 64 + c * 32);
        qa[c].u[0] = *(const v4u*)(base + half * 16);
        qa[c].u[1] = *(const v4u*)(base + half * 16 + 32);
    }

    v8f o[4];
    float mrow[8], lrow[8];
    #pragma unroll
    for (int i = 0; i < 8; ++i) { mrow[i] = -1e30f; lrow[i] = 0.0f; }
    #pragma unroll
    for (int s = 0; s < 4; ++s)
        #pragma unroll
        for (int i = 0; i < 8; ++i) o[s][i] = 0.0f;

    const int jmax = (qm + 47) >> 5;           // causal KV-chunk bound
    for (int j = 0; j < jmax; ++j) {
        const int kb = j * 32;

        // ---- scores: two 16x16 tiles over this 32-wide KV chunk ----
        v8f s0, s1;
        #pragma unroll
        for (int i = 0; i < 8; ++i) { s0[i] = 0.0f; s1[i] = 0.0f; }
        #pragma unroll
        for (int c = 0; c < 2; ++c) {
            FragBF bk0, bk1;    // B operand: lane = KV row, dims {0-7,16-23} per half
            const char* r0 = (const char*)(Kh + (size_t)(kb + l16) * 64)      + c * 64 + half * 16;
            const char* r1 = (const char*)(Kh + (size_t)(kb + 16 + l16) * 64) + c * 64 + half * 16;
            bk0.u[0] = *(const v4u*)(r0); bk0.u[1] = *(const v4u*)(r0 + 32);
            bk1.u[0] = *(const v4u*)(r1); bk1.u[1] = *(const v4u*)(r1 + 32);
            s0 = wmma_bf16(qa[c], bk0, s0);
            s1 = wmma_bf16(qa[c], bk1, s1);
        }

        // ---- causal mask (only near the diagonal) ----
        if (kb + 31 > qm) {
            #pragma unroll
            for (int i = 0; i < 8; ++i) {
                const int row = qm + i + 8 * half;
                const int c0  = kb + l16;
                if (c0 > row)      s0[i] = -1e30f;
                if (c0 + 16 > row) s1[i] = -1e30f;
            }
        }

        // ---- online softmax: row reductions across the 16-lane stripe ----
        #pragma unroll
        for (int i = 0; i < 8; ++i) {
            float mx = fmaxf(s0[i], s1[i]);
            mx = fmaxf(mx, __shfl_xor(mx, 1, 32));
            mx = fmaxf(mx, __shfl_xor(mx, 2, 32));
            mx = fmaxf(mx, __shfl_xor(mx, 4, 32));
            mx = fmaxf(mx, __shfl_xor(mx, 8, 32));
            const float mn    = fmaxf(mrow[i], mx);
            const float alpha = __expf(mrow[i] - mn);
            mrow[i] = mn;
            const float p0 = __expf(s0[i] - mn);
            const float p1 = __expf(s1[i] - mn);
            s0[i] = p0; s1[i] = p1;
            float rs = p0 + p1;
            rs += __shfl_xor(rs, 1, 32);
            rs += __shfl_xor(rs, 2, 32);
            rs += __shfl_xor(rs, 4, 32);
            rs += __shfl_xor(rs, 8, 32);
            lrow[i] = lrow[i] * alpha + rs;
            o[0][i] *= alpha; o[1][i] *= alpha; o[2][i] *= alpha; o[3][i] *= alpha;
        }

        // ---- transpose P (C layout -> A layout) through per-wave LDS ----
        #pragma unroll
        for (int i = 0; i < 8; ++i) {
            const int r = i + 8 * half;
            Ps[wave][r][l16]      = (bf16)s0[i];
            Ps[wave][r][16 + l16] = (bf16)s1[i];
        }
        asm volatile("s_wait_dscnt 0" ::: "memory");   // wave-local store->load order
        FragBF pa;
        {
            const char* pb = (const char*)&Ps[wave][l16][0];
            pa.u[0] = *(const v4u*)(pb + half * 16);
            pa.u[1] = *(const v4u*)(pb + half * 16 + 32);
        }

        // ---- O += P @ V : B operand columns are contiguous rows of V^T ----
        #pragma unroll
        for (int s = 0; s < 4; ++s) {
            FragBF bv;
            const char* vb = (const char*)(Vh + (size_t)(s * 16 + l16) * Nn + kb + half * 8);
            bv.u[0] = *(const v4u*)(vb);
            bv.u[1] = *(const v4u*)(vb + 32);
            o[s] = wmma_bf16(pa, bv, o[s]);
        }
    }

    // ---- normalize and scatter to O[token][1024] ----
    const int b = bh >> 4, h = bh & 15;
    #pragma unroll
    for (int i = 0; i < 8; ++i) {
        const float inv  = 1.0f / lrow[i];
        const int   qpos = qm + i + 8 * half;
        const size_t row = (size_t)(b * 2048 + qpos) * 1024;
        #pragma unroll
        for (int s = 0; s < 4; ++s) {
            const int col = h * 64 + s * 16 + l16;
            Op[row + col] = (bf16)(o[s][i] * inv);
        }
    }
}

// ---------------------------------------------------------------------------
extern "C" void kernel_launch(void* const* d_in, const int* in_sizes, int n_in,
                              void* d_out, int out_size, void* d_ws, size_t ws_size,
                              hipStream_t stream) {
    (void)in_sizes; (void)n_in; (void)out_size; (void)ws_size;
    const float* x  = (const float*)d_in[0];
    const float* wq = (const float*)d_in[1];
    const float* wk = (const float*)d_in[2];
    const float* wv = (const float*)d_in[3];
    const float* wo = (const float*)d_in[4];

    const int M = 4096, D = 1024;          // M = B*N tokens

    char* ws = (char*)d_ws;
    bf16* xb  = (bf16*)ws; ws += (size_t)M * D * sizeof(bf16);
    bf16* wqb = (bf16*)ws; ws += (size_t)D * D * sizeof(bf16);
    bf16* wkb = (bf16*)ws; ws += (size_t)D * D * sizeof(bf16);
    bf16* wvb = (bf16*)ws; ws += (size_t)D * D * sizeof(bf16);
    bf16* wob = (bf16*)ws; ws += (size_t)D * D * sizeof(bf16);
    bf16* Qb  = (bf16*)ws; ws += (size_t)M * D * sizeof(bf16);   // [b,h,n,64]
    bf16* Kb  = (bf16*)ws; ws += (size_t)M * D * sizeof(bf16);   // [b,h,n,64]
    bf16* Vtb = (bf16*)ws; ws += (size_t)M * D * sizeof(bf16);   // [b,h,64,n]
    bf16* Ob  = (bf16*)ws; ws += (size_t)M * D * sizeof(bf16);   // [token,1024]

    mha_cvt_bf16<<<(M * D / 4 + 255) / 256, 256, 0, stream>>>(x,  xb,  M * D / 4);
    mha_cvt_bf16<<<(D * D / 4 + 255) / 256, 256, 0, stream>>>(wq, wqb, D * D / 4);
    mha_cvt_bf16<<<(D * D / 4 + 255) / 256, 256, 0, stream>>>(wk, wkb, D * D / 4);
    mha_cvt_bf16<<<(D * D / 4 + 255) / 256, 256, 0, stream>>>(wv, wvb, D * D / 4);
    mha_cvt_bf16<<<(D * D / 4 + 255) / 256, 256, 0, stream>>>(wo, wob, D * D / 4);

    dim3 gGemm(D / 128, M / 128);          // 8 x 32 blocks
    mha_gemm_bf16<<<gGemm, 256, 0, stream>>>(xb, wqb, (void*)Qb,  D, 1);  // Q + RoPE + scale
    mha_gemm_bf16<<<gGemm, 256, 0, stream>>>(xb, wkb, (void*)Kb,  D, 2);  // K + RoPE
    mha_gemm_bf16<<<gGemm, 256, 0, stream>>>(xb, wvb, (void*)Vtb, D, 3);  // V transposed

    mha_flash_attn<<<dim3(2048 / 128, 32), 256, 0, stream>>>(Qb, Kb, Vtb, Ob);

    mha_gemm_bf16<<<gGemm, 256, 0, stream>>>(Ob, wob, d_out, D, 0);       // out-projection
}
